// GGNN_82403242541828
// MI455X (gfx1250) — matrix operand order repaired and checked by reading, not compiled, not run on hardware
//
#include <hip/hip_runtime.h>

// GGNN forward for MI455X (gfx1250, wave32, WMMA bf16).
// B=32, N=512, E=4, D=128, STEPS=5.
//
// Strategy: convert A (256MB f32) once to bf16 WMMA fragments (128MB -> fits
// in 192MB L2); all GEMMs via v_wmma_f32_16x16x32_bf16 with f32 accumulate.
// Fragment layouts match CDNA5 ISA 7.12.2:
//   A-operand tile (16Mx32K): lane L: M=L%16, h=L/16; elem i<8: K=i+8h,
//                             elem i>=8: K=16+(i-8)+8h  (two 8-wide runs)
//   B-operand tile (32Kx16N): lane L: N=L%16, h=L/16; same K order as A
//   C/D tile (16x16 f32):     acc[r]: row = r + 8*(L/16), col = L%16

#define GB 32
#define GN 512
#define GE 4
#define GD 128
#define GSTEPS 5

typedef __attribute__((ext_vector_type(16))) __bf16 v16bf;
typedef __attribute__((ext_vector_type(8)))  float  v8f;

static __device__ inline v8f wmma_bf16(v16bf a, v16bf b, v8f c) {
  return __builtin_amdgcn_wmma_f32_16x16x32_bf16(
      /*neg_a=*/false, a, /*neg_b=*/false, b,
      /*c_mod=*/(short)0, c, /*reuse_a=*/false, /*reuse_b=*/false);
}

static __device__ inline float sigmoidf_dev(float x) {
  return 1.0f / (1.0f + __expf(-x));
}

// ---------------------------------------------------------------------------
// One-time conversion: row-major f32 -> A-operand bf16 fragments.
// One wave per 16x32 tile; tile index = mt*numKt + kt (mt is global over B*N).
// ---------------------------------------------------------------------------
__global__ __launch_bounds__(256) void k_conv_afrag(
    const float* __restrict__ src, int rowStride, int colOff, int numKt,
    unsigned short* __restrict__ dst)
{
  int wave = (blockIdx.x * blockDim.x + threadIdx.x) >> 5;
  int lane = threadIdx.x & 31;
  int mt = wave / numKt, kt = wave - mt * numKt;
  int m = mt * 16 + (lane & 15);
  int h = lane >> 4;
  const float* s = src + (size_t)m * rowStride + colOff + kt * 32;
  v16bf f;
#pragma unroll
  for (int i = 0; i < 8; ++i) f[i] = (__bf16)s[8 * h + i];
#pragma unroll
  for (int i = 0; i < 8; ++i) f[8 + i] = (__bf16)s[16 + 8 * h + i];
  *(v16bf*)(dst + (size_t)wave * 512 + lane * 16) = f;
}

// ---------------------------------------------------------------------------
// One-time conversion: row-major f32 (K x Ncols) -> B-operand bf16 fragments.
// Tile index = kt*numNt + nt.
// ---------------------------------------------------------------------------
__global__ __launch_bounds__(256) void k_conv_bfrag(
    const float* __restrict__ src, int Ncols, int numNt,
    unsigned short* __restrict__ dst)
{
  int wave = (blockIdx.x * blockDim.x + threadIdx.x) >> 5;
  int lane = threadIdx.x & 31;
  int kt = wave / numNt, nt = wave - kt * numNt;
  int c = nt * 16 + (lane & 15);
  int kbase = kt * 32 + (lane >> 4) * 8;
  v16bf f;
#pragma unroll
  for (int i = 0; i < 16; ++i)
    f[i] = (__bf16)src[(size_t)(kbase + i + (i & 8)) * Ncols + c];
  *(v16bf*)(dst + (size_t)wave * 512 + lane * 16) = f;
}

// ---------------------------------------------------------------------------
// Expert GEMM: in/out_states[b, e*N+n, h] = sum_d state[b,n,d]*W[e,d,h] + bias
// Block = (b,e,n32): 32 rows. Wave w -> 16-col tile nt=w. Output written as
// B-operand fragments for the aggregation GEMM (32K x 16N tile == this tile).
// ---------------------------------------------------------------------------
__global__ __launch_bounds__(256) void k_expert(
    const unsigned short* __restrict__ stateF,   // A-frags [mtG*4 + kt]
    const unsigned short* __restrict__ inWf,     // [e*32 + kt*8 + nt]
    const unsigned short* __restrict__ outWf,
    const float* __restrict__ in_b, const float* __restrict__ out_b,  // (E,D)
    unsigned short* __restrict__ inSf,           // [(b*64+ktOut)*8 + nt]
    unsigned short* __restrict__ outSf)
{
  __shared__ float lds[8][512];  // per-wave 32x16 col-major [c*32 + row]
  int w = threadIdx.x >> 5, lane = threadIdx.x & 31;
  int blk = blockIdx.x;
  int n32 = blk & 15;
  int e = (blk >> 4) & 3;
  int b = blk >> 6;
  int mtA = b * 32 + n32 * 2;  // two 16-row tiles
  int c = lane & 15, half = lane >> 4;

  v8f ai0 = (v8f)0.0f, ai1 = (v8f)0.0f, ao0 = (v8f)0.0f, ao1 = (v8f)0.0f;
#pragma unroll
  for (int kt = 0; kt < 4; ++kt) {
    v16bf a0 = *(const v16bf*)(stateF + ((size_t)mtA * 4 + kt) * 512 + lane * 16);
    v16bf a1 = *(const v16bf*)(stateF + ((size_t)(mtA + 1) * 4 + kt) * 512 + lane * 16);
    v16bf bi = *(const v16bf*)(inWf  + ((size_t)e * 32 + kt * 8 + w) * 512 + lane * 16);
    v16bf bo = *(const v16bf*)(outWf + ((size_t)e * 32 + kt * 8 + w) * 512 + lane * 16);
    ai0 = wmma_bf16(a0, bi, ai0);
    ai1 = wmma_bf16(a1, bi, ai1);
    ao0 = wmma_bf16(a0, bo, ao0);
    ao1 = wmma_bf16(a1, bo, ao1);
  }
  float bin = in_b[e * GD + w * 16 + c];
  float bon = out_b[e * GD + w * 16 + c];
  int ktOut = e * 16 + n32;
  float* L = lds[w];

  // ---- in_states ----
#pragma unroll
  for (int r = 0; r < 8; ++r) L[c * 32 + 8 * half + r] = ai0[r] + bin;
#pragma unroll
  for (int r = 0; r < 8; ++r) L[c * 32 + 16 + 8 * half + r] = ai1[r] + bin;
  asm volatile("s_wait_dscnt 0" ::: "memory");
  {
    v16bf f;
#pragma unroll
    for (int i = 0; i < 16; ++i) f[i] = (__bf16)L[c * 32 + half * 8 + i + (i & 8)];
    *(v16bf*)(inSf + ((size_t)(b * 64 + ktOut) * 8 + w) * 512 + lane * 16) = f;
  }
  asm volatile("s_wait_dscnt 0" ::: "memory");
  // ---- out_states ----
#pragma unroll
  for (int r = 0; r < 8; ++r) L[c * 32 + 8 * half + r] = ao0[r] + bon;
#pragma unroll
  for (int r = 0; r < 8; ++r) L[c * 32 + 16 + 8 * half + r] = ao1[r] + bon;
  asm volatile("s_wait_dscnt 0" ::: "memory");
  {
    v16bf f;
#pragma unroll
    for (int i = 0; i < 16; ++i) f[i] = (__bf16)L[c * 32 + half * 8 + i + (i & 8)];
    *(v16bf*)(outSf + ((size_t)(b * 64 + ktOut) * 8 + w) * 512 + lane * 16) = f;
  }
}

// ---------------------------------------------------------------------------
// Aggregation GEMM: a_in[b] = A_in[b](512x2048) x in_states[b](2048x128),
// same for out. Block = (b*32+mt). Waves 0-3: a_in col-pairs; 4-7: a_out.
// Each wave: 16 rows x 32 cols -> one complete A-operand fragment tile.
// ---------------------------------------------------------------------------
__global__ __launch_bounds__(256) void k_agg(
    const unsigned short* __restrict__ Ainf,   // [(b*32+mt)*64 + kt]
    const unsigned short* __restrict__ Aoutf,
    const unsigned short* __restrict__ inSf,   // [(b*64+kt)*8 + nt]
    const unsigned short* __restrict__ outSf,
    unsigned short* __restrict__ aInf,         // [(b*32+mt)*4 + ktOut]
    unsigned short* __restrict__ aOutf)
{
  __shared__ float lds[8][512];  // per-wave 16x32 row-major [m*32 + k]
  int w = threadIdx.x >> 5, lane = threadIdx.x & 31;
  int blk = blockIdx.x;
  int isOut = w >> 2, ntp = w & 3;
  const unsigned short* Af = isOut ? Aoutf : Ainf;
  const unsigned short* Sf = isOut ? outSf : inSf;
  int b = blk >> 5;
  int c = lane & 15, half = lane >> 4;

  v8f acc0 = (v8f)0.0f, acc1 = (v8f)0.0f;
  size_t aBase = (size_t)blk * 64 * 512 + lane * 16;
  size_t sBase = ((size_t)b * 64 * 8 + ntp * 2) * 512 + lane * 16;
  for (int kt = 0; kt < 64; ++kt) {
    v16bf a  = *(const v16bf*)(Af + aBase + (size_t)kt * 512);
    v16bf b0 = *(const v16bf*)(Sf + sBase + (size_t)kt * 8 * 512);
    v16bf b1 = *(const v16bf*)(Sf + sBase + (size_t)(kt * 8 + 1) * 512);
    acc0 = wmma_bf16(a, b0, acc0);
    acc1 = wmma_bf16(a, b1, acc1);
  }
  float* L = lds[w];
#pragma unroll
  for (int r = 0; r < 8; ++r) L[(8 * half + r) * 32 + c] = acc0[r];
#pragma unroll
  for (int r = 0; r < 8; ++r) L[(8 * half + r) * 32 + 16 + c] = acc1[r];
  asm volatile("s_wait_dscnt 0" ::: "memory");
  v16bf f;
#pragma unroll
  for (int i = 0; i < 8; ++i) f[i] = (__bf16)L[c * 32 + 8 * half + i];
#pragma unroll
  for (int i = 0; i < 8; ++i) f[8 + i] = (__bf16)L[c * 32 + 16 + 8 * half + i];
  unsigned short* dst = isOut ? aOutf : aInf;
  *(v16bf*)(dst + ((size_t)blk * 4 + ntp) * 512 + lane * 16) = f;
}

// ---------------------------------------------------------------------------
// GRU update. Block = 16-row tile (mt in 0..1023). Wave w -> 16-col tile.
// a = [a_in | a_out | state] (K=384). Updates state f32 + bf16 A-frag mirror.
// ---------------------------------------------------------------------------
__global__ __launch_bounds__(256) void k_gru(
    const unsigned short* __restrict__ aInf,
    const unsigned short* __restrict__ aOutf,
    unsigned short* __restrict__ stateF,
    float* __restrict__ stateV,
    const unsigned short* __restrict__ rWf, const float* __restrict__ rb,
    const unsigned short* __restrict__ zWf, const float* __restrict__ zb,
    const unsigned short* __restrict__ tWf, const float* __restrict__ tb)
{
  __shared__ float lds[16 * 128];  // shared 16-row x 128-col f32 tile
  int w = threadIdx.x >> 5, lane = threadIdx.x & 31;
  int mt = blockIdx.x;
  int c = lane & 15, half = lane >> 4;
  int col = w * 16 + c;

  auto loadA = [&](int kt) -> v16bf {
    const unsigned short* p =
        (kt < 4) ? aInf + ((size_t)mt * 4 + kt) * 512
        : (kt < 8) ? aOutf + ((size_t)mt * 4 + (kt - 4)) * 512
                   : stateF + ((size_t)mt * 4 + (kt - 8)) * 512;
    return *(const v16bf*)(p + lane * 16);
  };

  // r gate
  v8f accR = (v8f)0.0f;
  for (int kt = 0; kt < 12; ++kt) {
    v16bf a = loadA(kt);
    v16bf bb = *(const v16bf*)(rWf + ((size_t)kt * 8 + w) * 512 + lane * 16);
    accR = wmma_bf16(a, bb, accR);
  }
  float rbv = rb[col];
  float s[8];
#pragma unroll
  for (int r = 0; r < 8; ++r) {
    int row = mt * 16 + 8 * half + r;
    s[r] = stateV[(size_t)row * GD + col];
    float rv = sigmoidf_dev(accR[r] + rbv);
    lds[(8 * half + r) * 128 + col] = rv * s[r];  // r * state
  }
  __syncthreads();

  // z gate
  v8f accZ = (v8f)0.0f;
  for (int kt = 0; kt < 12; ++kt) {
    v16bf a = loadA(kt);
    v16bf bb = *(const v16bf*)(zWf + ((size_t)kt * 8 + w) * 512 + lane * 16);
    accZ = wmma_bf16(a, bb, accZ);
  }

  // candidate h: joined = [a_in | a_out | r*state]
  v8f accH = (v8f)0.0f;
  for (int kt = 0; kt < 8; ++kt) {
    v16bf a = loadA(kt);
    v16bf bb = *(const v16bf*)(tWf + ((size_t)kt * 8 + w) * 512 + lane * 16);
    accH = wmma_bf16(a, bb, accH);
  }
  for (int kt = 8; kt < 12; ++kt) {
    v16bf a;
    int kb = (kt - 8) * 32 + 8 * half;
#pragma unroll
    for (int i = 0; i < 8; ++i) a[i] = (__bf16)lds[c * 128 + kb + i];
#pragma unroll
    for (int i = 0; i < 8; ++i) a[8 + i] = (__bf16)lds[c * 128 + kb + 16 + i];
    v16bf bb = *(const v16bf*)(tWf + ((size_t)kt * 8 + w) * 512 + lane * 16);
    accH = wmma_bf16(a, bb, accH);
  }

  float zbv = zb[col], tbv = tb[col];
  float sn[8];
#pragma unroll
  for (int r = 0; r < 8; ++r) {
    float z = sigmoidf_dev(accZ[r] + zbv);
    float h = tanhf(accH[r] + tbv);
    sn[r] = (1.0f - z) * s[r] + z * h;
  }
  __syncthreads();  // everyone done reading r*state from LDS
#pragma unroll
  for (int r = 0; r < 8; ++r) {
    int row = mt * 16 + 8 * half + r;
    stateV[(size_t)row * GD + col] = sn[r];
    lds[(8 * half + r) * 128 + col] = sn[r];
  }
  __syncthreads();
  if (w < 4) {  // rebuild bf16 A-operand fragments of the new state
    int kt = w;
    int m = lane & 15;
    int kb = kt * 32 + 8 * half;
    v16bf f;
#pragma unroll
    for (int i = 0; i < 8; ++i) f[i] = (__bf16)lds[m * 128 + kb + i];
#pragma unroll
    for (int i = 0; i < 8; ++i) f[8 + i] = (__bf16)lds[m * 128 + kb + 16 + i];
    *(v16bf*)(stateF + ((size_t)mt * 4 + kt) * 512 + lane * 16) = f;
  }
}

// ---------------------------------------------------------------------------
// Final: out = tanh([state | init_state] @ oW + ob), f32 row-major output.
// ---------------------------------------------------------------------------
__global__ __launch_bounds__(256) void k_final(
    const unsigned short* __restrict__ stateF,
    const unsigned short* __restrict__ initF,
    const unsigned short* __restrict__ oWf, const float* __restrict__ ob,
    float* __restrict__ out)
{
  int w = threadIdx.x >> 5, lane = threadIdx.x & 31;
  int mt = blockIdx.x;
  v8f acc = (v8f)0.0f;
  for (int kt = 0; kt < 8; ++kt) {
    const unsigned short* p = (kt < 4)
        ? stateF + ((size_t)mt * 4 + kt) * 512
        : initF + ((size_t)mt * 4 + (kt - 4)) * 512;
    v16bf a = *(const v16bf*)(p + lane * 16);
    v16bf bb = *(const v16bf*)(oWf + ((size_t)kt * 8 + w) * 512 + lane * 16);
    acc = wmma_bf16(a, bb, acc);
  }
  int c = lane & 15, half = lane >> 4, col = w * 16 + c;
  float obv = ob[col];
#pragma unroll
  for (int r = 0; r < 8; ++r) {
    int row = mt * 16 + 8 * half + r;
    out[(size_t)row * GD + col] = tanhf(acc[r] + obv);
  }
}

// ---------------------------------------------------------------------------
extern "C" void kernel_launch(void* const* d_in, const int* in_sizes, int n_in,
                              void* d_out, int out_size, void* d_ws, size_t ws_size,
                              hipStream_t stream) {
  (void)in_sizes; (void)n_in; (void)out_size; (void)ws_size;
  const float* prop  = (const float*)d_in[0];
  const float* A     = (const float*)d_in[1];
  /* d_in[2] node_mask unused */
  const float* in_W  = (const float*)d_in[3];
  const float* in_b  = (const float*)d_in[4];
  const float* out_W = (const float*)d_in[5];
  const float* out_b = (const float*)d_in[6];
  const float* rW = (const float*)d_in[7];  const float* rb = (const float*)d_in[8];
  const float* zW = (const float*)d_in[9];  const float* zb = (const float*)d_in[10];
  const float* tW = (const float*)d_in[11]; const float* tb = (const float*)d_in[12];
  const float* oW = (const float*)d_in[13]; const float* ob = (const float*)d_in[14];

  char* ws = (char*)d_ws;
  size_t off = 0;
  auto wsAlloc = [&](size_t bytes) -> char* {
    char* p = ws + off;
    off += (bytes + 255) & ~(size_t)255;
    return p;
  };
  // fragment tiles are 1024 B (512 ushort)
  unsigned short* Ainf   = (unsigned short*)wsAlloc((size_t)GB * 32 * 64 * 1024);  // 64 MB
  unsigned short* Aoutf  = (unsigned short*)wsAlloc((size_t)GB * 32 * 64 * 1024);  // 64 MB
  unsigned short* inSf   = (unsigned short*)wsAlloc((size_t)GB * 64 * 8 * 1024);   // 16 MB
  unsigned short* outSf  = (unsigned short*)wsAlloc((size_t)GB * 64 * 8 * 1024);   // 16 MB
  unsigned short* aInf   = (unsigned short*)wsAlloc((size_t)1024 * 4 * 1024);      //  4 MB
  unsigned short* aOutf  = (unsigned short*)wsAlloc((size_t)1024 * 4 * 1024);      //  4 MB
  unsigned short* stateF = (unsigned short*)wsAlloc((size_t)1024 * 4 * 1024);      //  4 MB
  unsigned short* initF  = (unsigned short*)wsAlloc((size_t)1024 * 4 * 1024);      //  4 MB
  unsigned short* rWf    = (unsigned short*)wsAlloc(96 * 1024);
  unsigned short* zWf    = (unsigned short*)wsAlloc(96 * 1024);
  unsigned short* tWf    = (unsigned short*)wsAlloc(96 * 1024);
  unsigned short* oWf    = (unsigned short*)wsAlloc(64 * 1024);
  unsigned short* inWf   = (unsigned short*)wsAlloc((size_t)GE * 32 * 1024);
  unsigned short* outWf  = (unsigned short*)wsAlloc((size_t)GE * 32 * 1024);
  float* stateV          = (float*)wsAlloc((size_t)GB * GN * GD * sizeof(float));  //  8 MB

  // --- one-time conversions (8 waves per 256-thread block; tiles % 8 == 0) ---
  // A_in / A_out: 1024 mt-tiles x 64 kt-tiles = 65536 tiles each
  k_conv_afrag<<<65536 / 8, 256, 0, stream>>>(A, 2 * GN * GE, 0,        64, Ainf);
  k_conv_afrag<<<65536 / 8, 256, 0, stream>>>(A, 2 * GN * GE, GN * GE,  64, Aoutf);
  // prop_state -> init frags and live-state frags (1024 x 4 = 4096 tiles)
  k_conv_afrag<<<4096 / 8, 256, 0, stream>>>(prop, GD, 0, 4, initF);
  k_conv_afrag<<<4096 / 8, 256, 0, stream>>>(prop, GD, 0, 4, stateF);
  hipMemcpyAsync(stateV, prop, (size_t)GB * GN * GD * sizeof(float),
                 hipMemcpyDeviceToDevice, stream);
  // weights
  k_conv_bfrag<<<96 / 8, 256, 0, stream>>>(rW, GD, 8, rWf);   // 12 kt x 8 nt
  k_conv_bfrag<<<96 / 8, 256, 0, stream>>>(zW, GD, 8, zWf);
  k_conv_bfrag<<<96 / 8, 256, 0, stream>>>(tW, GD, 8, tWf);
  k_conv_bfrag<<<64 / 8, 256, 0, stream>>>(oW, GD, 8, oWf);   //  8 kt x 8 nt
  for (int e = 0; e < GE; ++e) {                              //  4 kt x 8 nt per expert
    k_conv_bfrag<<<32 / 8, 256, 0, stream>>>(in_W + (size_t)e * GD * GD, GD, 8,
                                             inWf + (size_t)e * 32 * 512);
    k_conv_bfrag<<<32 / 8, 256, 0, stream>>>(out_W + (size_t)e * GD * GD, GD, 8,
                                             outWf + (size_t)e * 32 * 512);
  }

  // --- propagation steps ---
  for (int step = 0; step < GSTEPS; ++step) {
    k_expert<<<GB * GE * 16, 256, 0, stream>>>(stateF, inWf, outWf, in_b, out_b,
                                               inSf, outSf);
    k_agg<<<GB * 32, 256, 0, stream>>>(Ainf, Aoutf, inSf, outSf, aInf, aOutf);
    k_gru<<<1024, 256, 0, stream>>>(aInf, aOutf, stateF, stateV,
                                    rWf, rb, zWf, zb, tWf, tb);
  }

  // --- output ---
  k_final<<<1024, 256, 0, stream>>>(stateF, initF, oWf, ob, (float*)d_out);
}
